// StreamingErbDecoder_59545426591948
// MI455X (gfx1250) — matrix operand order repaired and checked by reading, not compiled, not run on hardware
//
#include <hip/hip_runtime.h>
#include <hip/hip_bf16.h>
#include <math.h>

// ---------------------------------------------------------------------------
// CDNA5 / gfx1250: wave32, WMMA 16x16x32 bf16 (f32 accumulate), TDM tile loads.
// ---------------------------------------------------------------------------

typedef __attribute__((ext_vector_type(16))) __bf16 v16bf;
typedef __attribute__((ext_vector_type(8)))  float  v8f;
typedef __attribute__((ext_vector_type(4)))  unsigned int v4u;
typedef __attribute__((ext_vector_type(8)))  int  v8i;
typedef __attribute__((ext_vector_type(4)))  int  v4i;

__device__ __forceinline__ v8f wmma_bf16(v16bf a, v16bf b, v8f c) {
  // (neg_a, A, neg_b, B, c_mod, C, reuse_a, reuse_b)
  return __builtin_amdgcn_wmma_f32_16x16x32_bf16(false, a, false, b, (short)0, c,
                                                 false, false);
}

// ---------------------------------------------------------------------------
// Tensor Data Mover: load a 2D tile (tile_w contiguous elems x tile_h rows of
// f32, row pitch row_stride elems) from global into LDS at lds_off.
// tensor_w bounds the contiguous dimension for OOB zero-fill (ISA D# §8).
// ---------------------------------------------------------------------------
__device__ __forceinline__ void tdm_load_2d(unsigned lds_off, const void* gaddr,
                                            unsigned tile_w, unsigned tile_h,
                                            unsigned row_stride, unsigned tensor_w) {
  unsigned long long ga = (unsigned long long)(uintptr_t)gaddr;
  v4u g0;
  g0[0] = 1u;                                        // count=1 (user descriptor)
  g0[1] = lds_off;                                   // LDS byte address
  g0[2] = (unsigned)(ga & 0xffffffffu);              // global_addr[31:0]
  g0[3] = (unsigned)((ga >> 32) & 0x01ffffffu)       // global_addr[56:32]
          | (2u << 30);                              // type=2 ("image")
  v8i g1;
  g1[0] = (int)(2u << 16);                           // data_size=2 (4 bytes)
  g1[1] = (int)((tensor_w & 0xffffu) << 16);         // tensor_dim0[15:0]
  g1[2] = (int)((tensor_w >> 16) & 0xffffu)          // tensor_dim0[31:16]
          | (int)((tile_h & 0xffffu) << 16);         // tensor_dim1[15:0]
  g1[3] = (int)((tile_w & 0xffffu) << 16);           // tensor_dim1[31:16]=0 | tile_dim0
  g1[4] = (int)(tile_h & 0xffffu);                   // tile_dim1 | tile_dim2=0
  g1[5] = (int)row_stride;                           // tensor_dim0_stride[31:0]
  g1[6] = 0;                                         // stride[47:32] | dim1_stride lo
  g1[7] = 0;
  v4i z4 = {0, 0, 0, 0};
  v8i z8 = {0, 0, 0, 0, 0, 0, 0, 0};
  __builtin_amdgcn_tensor_load_to_lds(g0, g1, z4, z4, z8, 0);
  __builtin_amdgcn_s_wait_tensorcnt(0);
}

// A fragment: 16x32 (MxK). lanes 0-15: row m0+lane, K = k0+{0..7,16..23}
//             lanes 16-31: row m0+lane-16, K = k0+{8..15,24..31}
__device__ __forceinline__ v16bf a_frag_rowmajor(const float* __restrict__ W,
                                                 int ld, int m0, int k0, int lane) {
  int m  = m0 + (lane & 15);
  int kb = k0 + ((lane & 16) ? 8 : 0);
  const float* p = W + (size_t)m * ld + kb;
  float4 l0 = *(const float4*)(p + 0);
  float4 l1 = *(const float4*)(p + 4);
  float4 h0 = *(const float4*)(p + 16);
  float4 h1 = *(const float4*)(p + 20);
  v16bf a;
  a[0]=(__bf16)l0.x;  a[1]=(__bf16)l0.y;  a[2]=(__bf16)l0.z;  a[3]=(__bf16)l0.w;
  a[4]=(__bf16)l1.x;  a[5]=(__bf16)l1.y;  a[6]=(__bf16)l1.z;  a[7]=(__bf16)l1.w;
  a[8]=(__bf16)h0.x;  a[9]=(__bf16)h0.y;  a[10]=(__bf16)h0.z; a[11]=(__bf16)h0.w;
  a[12]=(__bf16)h1.x; a[13]=(__bf16)h1.y; a[14]=(__bf16)h1.z; a[15]=(__bf16)h1.w;
  return a;
}

// A fragment with strided elements (ConvTranspose weights (O,C,1,3) at tap k).
__device__ __forceinline__ v16bf a_frag_strided(const float* __restrict__ Wk,
                                                int rowpitch, int estride,
                                                int m0, int k0, int lane) {
  int m  = m0 + (lane & 15);
  int kb = k0 + ((lane & 16) ? 8 : 0);
  const float* p = Wk + (size_t)m * rowpitch;
  v16bf a;
  #pragma unroll
  for (int i = 0; i < 8; ++i) a[i]     = (__bf16)p[(size_t)(kb + i) * estride];
  #pragma unroll
  for (int i = 0; i < 8; ++i) a[8 + i] = (__bf16)p[(size_t)(kb + 16 + i) * estride];
  return a;
}

// B fragment: 32x16 (KxN), column K-contiguous in global memory. Branch-free:
// callers guarantee col points at readable data (invalid columns may load
// duplicate data; the corresponding accumulator columns are never stored).
__device__ __forceinline__ v16bf b_frag_colmajor(const float* __restrict__ col,
                                                 int k0, int lane) {
  int kb = k0 + ((lane & 16) ? 8 : 0);
  const float4* p = (const float4*)(col + kb);
  float4 v0 = p[0], v1 = p[1], v2 = p[4], v3 = p[5];
  v16bf b;
  b[0]=(__bf16)v0.x;  b[1]=(__bf16)v0.y;  b[2]=(__bf16)v0.z;  b[3]=(__bf16)v0.w;
  b[4]=(__bf16)v1.x;  b[5]=(__bf16)v1.y;  b[6]=(__bf16)v1.z;  b[7]=(__bf16)v1.w;
  b[8]=(__bf16)v2.x;  b[9]=(__bf16)v2.y;  b[10]=(__bf16)v2.z; b[11]=(__bf16)v2.w;
  b[12]=(__bf16)v3.x; b[13]=(__bf16)v3.y; b[14]=(__bf16)v3.z; b[15]=(__bf16)v3.w;
  return b;
}

// B fragment gathered from an LDS tile (rows = K channels, ldt columns).
__device__ __forceinline__ v16bf b_frag_lds(const float* t, int ldt, int k0,
                                            int colofs, bool masked, int lane) {
  int col = (lane & 15) + colofs;
  int kb  = k0 + ((lane & 16) ? 8 : 0);
  v16bf b;
  #pragma unroll
  for (int i = 0; i < 16; ++i)
    b[i] = masked ? (__bf16)0.0f : (__bf16)t[(kb + i + ((i & 8) ? 8 : 0)) * ldt + col];
  return b;
}

__device__ __forceinline__ float sigf(float x) { return 1.0f / (1.0f + __expf(-x)); }

// ---------------------------------------------------------------------------
// Kernel 1: xg = emb @ Wih^T + bih   (M=3072, N=4096(b*t), K=1024)
// ---------------------------------------------------------------------------
__global__ void k_xg(const float* __restrict__ emb, const float* __restrict__ Wih,
                     const float* __restrict__ bih, float* __restrict__ xg) {
  int lane = threadIdx.x;
  int n0 = blockIdx.x * 16;
  int m0 = blockIdx.y * 16;
  const float* col = emb + (size_t)(n0 + (lane & 15)) * 1024;
  v8f acc = {};
  #pragma unroll 1
  for (int k0 = 0; k0 < 1024; k0 += 32) {
    v16bf a = a_frag_rowmajor(Wih, 1024, m0, k0, lane);
    v16bf b = b_frag_colmajor(col, k0, lane);
    acc = wmma_bf16(a, b, acc);
  }
  int n  = n0 + (lane & 15);
  int mr = m0 + ((lane & 16) ? 8 : 0);
  #pragma unroll
  for (int r = 0; r < 8; ++r) {
    int m = mr + r;
    xg[(size_t)n * 3072 + m] = acc[r] + bih[m];
  }
}

// ---------------------------------------------------------------------------
// Kernel 2: persistent GRU; 64 single-wave blocks, device-scope step barrier.
// Columns 8-15 of each tile duplicate batches 0-7 (branch-free B fragments);
// their results are discarded at the epilogue.
// ---------------------------------------------------------------------------
__global__ void k_gru(const float* __restrict__ xg, const float* __restrict__ Whh,
                      const float* __restrict__ bhh, float* __restrict__ hbuf,
                      float* __restrict__ emb2, unsigned* __restrict__ ctr) {
  int lane = threadIdx.x;
  int j0   = blockIdx.x * 16;
  int nlo  = lane & 15;
  int hi8  = (lane & 16) ? 8 : 0;
  size_t coloff = (size_t)(nlo & 7) * 1024;   // lanes 8-15 mirror batches 0-7

  #pragma unroll 1
  for (int s = 0; s < 512; ++s) {
    const float* cur = hbuf + (size_t)(s & 1) * 8192;
    float*       nxt = hbuf + (size_t)((s + 1) & 1) * 8192;
    const float* col = cur + coloff;

    v8f ar = {}, az = {}, an = {};
    #pragma unroll 1
    for (int k0 = 0; k0 < 1024; k0 += 32) {
      v16bf b  = b_frag_colmajor(col, k0, lane);
      v16bf wr = a_frag_rowmajor(Whh, 1024, j0,        k0, lane);
      v16bf wz = a_frag_rowmajor(Whh, 1024, 1024 + j0, k0, lane);
      v16bf wn = a_frag_rowmajor(Whh, 1024, 2048 + j0, k0, lane);
      ar = wmma_bf16(wr, b, ar);
      az = wmma_bf16(wz, b, az);
      an = wmma_bf16(wn, b, an);
    }

    if (nlo < 8) {
      int bb = nlo;
      const float* xgp = xg + ((size_t)bb * 512 + (size_t)s) * 3072;
      #pragma unroll
      for (int r = 0; r < 8; ++r) {
        int j = j0 + hi8 + r;
        float hr = ar[r] + bhh[j];
        float hz = az[r] + bhh[1024 + j];
        float hn = an[r] + bhh[2048 + j];
        float rg = sigf(xgp[j] + hr);
        float zg = sigf(xgp[1024 + j] + hz);
        float ng = tanhf(xgp[2048 + j] + rg * hn);
        float hp = cur[(size_t)bb * 1024 + j];
        float hv = (1.0f - zg) * ng + zg * hp;
        nxt[(size_t)bb * 1024 + j] = hv;
        int c = j & 127, f = j >> 7;
        emb2[(((size_t)bb * 128 + c) * 512 + (size_t)s) * 8 + f] = hv;
      }
    }

    __builtin_amdgcn_fence(__ATOMIC_RELEASE, "agent");
    if (lane == 0) atomicAdd(ctr, 1u);
    unsigned target = (unsigned)(s + 1) * 64u;
    while (*(volatile unsigned*)ctr < target) { }
    __builtin_amdgcn_fence(__ATOMIC_ACQUIRE, "agent");
  }
}

// ---------------------------------------------------------------------------
// Kernel 3: 1x1 conv + bias + skip. 8 waves/WG cover all 128 out channels;
// the (128ch x 16pos) activation tile comes in once via the TDM.
// ---------------------------------------------------------------------------
__global__ void __launch_bounds__(256)
k_pconv(const float* __restrict__ x, const float* __restrict__ W,
        const float* __restrict__ bias, const float* __restrict__ skip,
        float* __restrict__ out, int TF) {
  __shared__ float tile[128 * 16];
  int lane = threadIdx.x & 31;
  int wave = threadIdx.x >> 5;
  int p0 = blockIdx.x * 16;
  int bb = blockIdx.z;
  const float* xb = x + (size_t)bb * 128 * (size_t)TF + p0;

  if (threadIdx.x < 32) {
    tdm_load_2d((unsigned)(uintptr_t)tile, xb, 16u, 128u,
                (unsigned)TF, (unsigned)(TF - p0));
  }
  __syncthreads();

  int m0 = wave * 16;
  v8f acc = {};
  #pragma unroll 1
  for (int k0 = 0; k0 < 128; k0 += 32) {
    v16bf a = a_frag_rowmajor(W, 128, m0, k0, lane);
    v16bf b = b_frag_lds(tile, 16, k0, 0, false, lane);
    acc = wmma_bf16(a, b, acc);
  }
  int n  = p0 + (lane & 15);
  int mr = m0 + ((lane & 16) ? 8 : 0);
  #pragma unroll
  for (int r = 0; r < 8; ++r) {
    int m = mr + r;
    size_t idx = ((size_t)bb * 128 + m) * (size_t)TF + n;
    out[idx] = acc[r] + bias[m] + skip[idx];
  }
}

// ---------------------------------------------------------------------------
// Kernel 4: ConvTranspose2d k=(1,3), s=(1,2), p=(0,1), op=(0,1) + ReLU.
//   y[..,2q]   = W1 @ x[..,q]
//   y[..,2q+1] = W0 @ x[..,q] + W2 @ x[..,q+1]   (masked at q==F-1)
// One 17-column TDM tile serves both the base and the shifted fragments;
// tensor_dim0 clamp zero-fills the final overrun column.
// ---------------------------------------------------------------------------
__global__ void __launch_bounds__(256)
k_convt(const float* __restrict__ x, const float* __restrict__ wt,
        const float* __restrict__ bias, float* __restrict__ out, int F) {
  __shared__ float tile[128 * 17];
  int lane = threadIdx.x & 31;
  int wave = threadIdx.x >> 5;
  int p0 = blockIdx.x * 16;
  int bb = blockIdx.z;
  int TF = 512 * F;
  const float* xb = x + (size_t)bb * 128 * (size_t)TF + p0;

  if (threadIdx.x < 32) {
    tdm_load_2d((unsigned)(uintptr_t)tile, xb, 17u, 128u,
                (unsigned)TF, (unsigned)(TF - p0));
  }
  __syncthreads();

  int m0 = wave * 16;
  int n  = p0 + (lane & 15);
  int q  = n % F;
  bool mask_s = (q == F - 1);
  v8f accE = {}, accO = {};
  #pragma unroll 1
  for (int k0 = 0; k0 < 128; k0 += 32) {
    v16bf b  = b_frag_lds(tile, 17, k0, 0, false, lane);
    v16bf bs = b_frag_lds(tile, 17, k0, 1, mask_s, lane);
    v16bf a1 = a_frag_strided(wt + 1, 384, 3, m0, k0, lane);
    v16bf a0 = a_frag_strided(wt + 0, 384, 3, m0, k0, lane);
    v16bf a2 = a_frag_strided(wt + 2, 384, 3, m0, k0, lane);
    accE = wmma_bf16(a1, b,  accE);
    accO = wmma_bf16(a0, b,  accO);
    accO = wmma_bf16(a2, bs, accO);
  }
  int t  = n / F;
  int mr = m0 + ((lane & 16) ? 8 : 0);
  #pragma unroll
  for (int r = 0; r < 8; ++r) {
    int m = mr + r;
    float bv = bias[m];
    size_t base = (((size_t)bb * 128 + m) * 512 + t) * (size_t)(2 * F);
    float e = accE[r] + bv;
    float o = accO[r] + bv;
    out[base + 2 * q]     = e > 0.0f ? e : 0.0f;
    out[base + 2 * q + 1] = o > 0.0f ? o : 0.0f;
  }
}

// ---------------------------------------------------------------------------
// Kernel 5: final 3-tap conv over freq (128 -> 1 channel) + sigmoid.
// ---------------------------------------------------------------------------
__global__ void k_mask(const float* __restrict__ pre, const float* __restrict__ w0o,
                       const float* __restrict__ b0o, float* __restrict__ out) {
  int idx = blockIdx.x * blockDim.x + threadIdx.x;   // 8*512*64
  if (idx >= 8 * 512 * 64) return;
  int f  = idx & 63;
  int t  = (idx >> 6) & 511;
  int bb = idx >> 15;
  float s = b0o[0];
  const float* pb = pre + (((size_t)bb * 128) * 512 + (size_t)t) * 64 + f;
  #pragma unroll 4
  for (int c = 0; c < 128; ++c) {
    const float* pc = pb + (size_t)c * 512 * 64;
    float w0 = w0o[c * 3 + 0], w1 = w0o[c * 3 + 1], w2 = w0o[c * 3 + 2];
    if (f > 0)  s += w0 * pc[-1];
    s += w1 * pc[0];
    if (f < 63) s += w2 * pc[1];
  }
  out[idx] = sigf(s);
}

__global__ void k_zero_ctr(unsigned* __restrict__ p) {
  if (threadIdx.x == 0) *p = 0u;
}

// ---------------------------------------------------------------------------
extern "C" void kernel_launch(void* const* d_in, const int* in_sizes, int n_in,
                              void* d_out, int out_size, void* d_ws, size_t ws_size,
                              hipStream_t stream) {
  const float* emb  = (const float*)d_in[0];
  const float* e3   = (const float*)d_in[1];
  const float* e2   = (const float*)d_in[2];
  const float* e1   = (const float*)d_in[3];
  const float* e0   = (const float*)d_in[4];
  const float* herb = (const float*)d_in[5];
  const float* Wih  = (const float*)d_in[6];
  const float* Whh  = (const float*)d_in[7];
  const float* bih  = (const float*)d_in[8];
  const float* bhh  = (const float*)d_in[9];
  const float* w3p = (const float*)d_in[10]; const float* b3p = (const float*)d_in[11];
  const float* w2p = (const float*)d_in[12]; const float* b2p = (const float*)d_in[13];
  const float* w1p = (const float*)d_in[14]; const float* b1p = (const float*)d_in[15];
  const float* w0p = (const float*)d_in[16]; const float* b0p = (const float*)d_in[17];
  const float* wt3 = (const float*)d_in[18]; const float* bt3 = (const float*)d_in[19];
  const float* wt2 = (const float*)d_in[20]; const float* bt2 = (const float*)d_in[21];
  const float* wt1 = (const float*)d_in[22]; const float* bt1 = (const float*)d_in[23];
  const float* w0o = (const float*)d_in[24]; const float* b0o = (const float*)d_in[25];

  float* ws = (float*)d_ws;
  const size_t OFF_XG  = 0;                          // 4096*3072
  const size_t OFF_H   = OFF_XG + (size_t)4096 * 3072;
  const size_t OFF_CTR = OFF_H + 2 * 8192;
  const size_t OFF_B0  = OFF_CTR + 64;               // 8*128*512*64
  const size_t OFF_B1  = OFF_B0 + (size_t)8 * 128 * 512 * 64;
  float* xg     = ws + OFF_XG;
  float* hbuf   = ws + OFF_H;
  unsigned* ctr = (unsigned*)(ws + OFF_CTR);
  float* buf0   = ws + OFF_B0;
  float* buf1   = ws + OFF_B1;
  float* outF   = (float*)d_out;

  k_zero_ctr<<<1, 32, 0, stream>>>(ctr);
  (void)hipMemcpyAsync(hbuf, herb, (size_t)8 * 1024 * sizeof(float),
                       hipMemcpyDeviceToDevice, stream);

  k_xg<<<dim3(256, 192), 32, 0, stream>>>(emb, Wih, bih, xg);
  k_gru<<<64, 32, 0, stream>>>(xg, Whh, bhh, hbuf, buf0, ctr);

  k_pconv<<<dim3(256,  1, 8), 256, 0, stream>>>(e3, w3p, b3p, buf0, buf1, 512 * 8);
  k_convt<<<dim3(256,  1, 8), 256, 0, stream>>>(buf1, wt3, bt3, buf0, 8);

  k_pconv<<<dim3(512,  1, 8), 256, 0, stream>>>(e2, w2p, b2p, buf0, buf1, 512 * 16);
  k_convt<<<dim3(512,  1, 8), 256, 0, stream>>>(buf1, wt2, bt2, buf0, 16);

  k_pconv<<<dim3(1024, 1, 8), 256, 0, stream>>>(e1, w1p, b1p, buf0, buf1, 512 * 32);
  k_convt<<<dim3(1024, 1, 8), 256, 0, stream>>>(buf1, wt1, bt1, buf0, 32);

  k_pconv<<<dim3(2048, 1, 8), 256, 0, stream>>>(e0, w0p, b0p, buf0, buf1, 512 * 64);

  k_mask<<<dim3((8 * 512 * 64 + 255) / 256), 256, 0, stream>>>(buf1, w0o, b0o, outF);

  (void)hipMemcpyAsync(outF + (size_t)8 * 512 * 64, hbuf,
                       (size_t)8 * 1024 * sizeof(float), hipMemcpyDeviceToDevice, stream);
}
